// Attention_86818468921452
// MI455X (gfx1250) — matrix operand order repaired and checked by reading, not compiled, not run on hardware
//
#include <hip/hip_runtime.h>

#define NTOK 8192
#define DIM  1024
#define INV_SCALE 0.08838834764831843f   // 1/sqrt(128)
#define PSTRIDE (NTOK + 8)               // LDS row pad: stride % 64 banks == 4
#define QSTRIDE (DIM + 8)

typedef __attribute__((ext_vector_type(16))) _Float16 v16h;
typedef __attribute__((ext_vector_type(8)))  _Float16 v8h;
typedef __attribute__((ext_vector_type(8)))  float    v8f;

// scheduler pipeline masks (LLVM AMDGPU sched_group_barrier)
#define SG_WMMA   0x008   // MFMA/WMMA group
#define SG_VMEMRD 0x020   // VMEM read
#define SG_DSRD   0x100   // LDS read

// ---- WMMA fragment helpers (CDNA5 ISA 7.12.2, 16-bit layouts, wave32) ----

// A-matrix 16x32 f16: lane L holds row M=L%16; this code's k order (not the ISA table): half=L/16 selects K-chunk
// [half*16, half*16+16) contiguous  -> two 16B loads.
__device__ __forceinline__ v16h load_a16(const _Float16* row_ptr, int k0, int half) {
  v8h a0 = *(const v8h*)(row_ptr + k0 + half * 16);
  v8h a1 = *(const v8h*)(row_ptr + k0 + half * 16 + 8);
  v16h r;
#pragma unroll
  for (int i = 0; i < 8; ++i) { r[i] = a0[i]; r[i + 8] = a1[i]; }
  return r;
}

// B-matrix 32x16 f16: lane L holds column N=L%16; same k order as A: K = [half*16, half*16+16)
// contiguous when the column is stored K-major (i.e. B^T rows) -> one 32B load.
__device__ __forceinline__ v16h load_b16(const _Float16* col_ptr, int k0, int half) {
  return *(const v16h*)(col_ptr + k0 + half * 16);
}

__device__ __forceinline__ v8f wmma_f16(v16h a, v16h b, v8f c) {
  return __builtin_amdgcn_wmma_f32_16x16x32_f16(false, a, false, b, (short)0, c,
                                                false, false);
}

// ---- fp32 -> f16 conversion (x and weights) ----
__global__ void cvt_f32_f16(const float* __restrict__ in, _Float16* __restrict__ out,
                            int n) {
  int i = blockIdx.x * blockDim.x + threadIdx.x;
  int stride = gridDim.x * blockDim.x;
  for (; i < n; i += stride) out[i] = (_Float16)in[i];
}

// ---- projection GEMM: out[n,d] = sum_k A[n,k]*W[d,k] + b[d] (f16 out) ----
// Each wave computes a 16x64 tile; A+B fragments are register double-buffered
// and the scheduler is pinned to "10 loads -> 4 WMMAs" per K-step.
__global__ void __launch_bounds__(256)
gemm_proj(const _Float16* __restrict__ A,   // [NTOK, DIM]
          const _Float16* __restrict__ W,   // [DIM, DIM] (row d = B-column d)
          const float* __restrict__ bias,   // [DIM]
          _Float16* __restrict__ out,       // q/k: [NTOK,DIM]; v: vT [DIM,NTOK]
          int store_transposed) {
  const int wave = threadIdx.x >> 5;
  const int lane = threadIdx.x & 31;
  const int half = lane >> 4;
  const int ln   = lane & 15;
  const int gw = blockIdx.x * 8 + wave;       // 8192 wave-tiles total
  const int m0 = (gw >> 4) << 4;              // 512 row-tiles
  const int d0 = (gw & 15) << 6;              // 16 col-groups of 64

  const _Float16* arow = A + (size_t)(m0 + ln) * DIM;
  const _Float16* wrow[4];
#pragma unroll
  for (int t = 0; t < 4; ++t) wrow[t] = W + (size_t)(d0 + t * 16 + ln) * DIM;

  v8f acc[4] = {};
  v16h a = load_a16(arow, 0, half);
  v16h b[4];
#pragma unroll
  for (int t = 0; t < 4; ++t) b[t] = load_b16(wrow[t], 0, half);

#pragma unroll
  for (int kk = 0; kk < DIM; kk += 32) {
    const int kn = kk + 32;
    v16h an = a;
    v16h bn[4] = {b[0], b[1], b[2], b[3]};
    if (kn < DIM) {                 // resolved statically after unroll
      an = load_a16(arow, kn, half);
#pragma unroll
      for (int t = 0; t < 4; ++t) bn[t] = load_b16(wrow[t], kn, half);
    }
#pragma unroll
    for (int t = 0; t < 4; ++t) acc[t] = wmma_f16(a, b[t], acc[t]);
    a = an;
#pragma unroll
    for (int t = 0; t < 4; ++t) b[t] = bn[t];
    // pipeline: issue next-step load clause, then this step's WMMA group
    __builtin_amdgcn_sched_group_barrier(SG_VMEMRD, 10, 0);
    __builtin_amdgcn_sched_group_barrier(SG_WMMA,    4, 0);
  }

#pragma unroll
  for (int t = 0; t < 4; ++t) {
    const int col = d0 + t * 16 + ln;
    const float bv = bias[col];
#pragma unroll
    for (int r = 0; r < 8; ++r) {
      const int row = m0 + half * 8 + r;
      const float v = acc[t][r] + bv;
      if (store_transposed) out[(size_t)col * NTOK + row] = (_Float16)v;
      else                  out[(size_t)row * DIM + col]  = (_Float16)v;
    }
  }
}

// ---- flash-style attention: 16 query rows / workgroup ----
// Q block (32KB) staged in LDS via CDNA5 async global->LDS; probabilities P
// (16 x 8192 f16, ~256KB) staged in the 320KB WGP LDS.
__global__ void __launch_bounds__(256)
flash_attn(const _Float16* __restrict__ q,   // [NTOK, DIM]
           const _Float16* __restrict__ k,   // [NTOK, DIM]
           const _Float16* __restrict__ vT,  // [DIM, NTOK]
           _Float16* __restrict__ o) {       // [NTOK, DIM]
  extern __shared__ __align__(16) char smem[];
  _Float16* P  = (_Float16*)smem;                                   // [16][PSTRIDE]
  _Float16* Qs = (_Float16*)(smem + (size_t)16 * PSTRIDE * 2);      // [16][QSTRIDE]
  float* mbuf  = (float*)(smem + (size_t)16 * PSTRIDE * 2 + (size_t)16 * QSTRIDE * 2);
  float* lbuf  = mbuf + 8 * 16;

  const int wave = threadIdx.x >> 5;
  const int lane = threadIdx.x & 31;
  const int half = lane >> 4;
  const int ln   = lane & 15;
  const int m0   = blockIdx.x * 16;
  const float NEG_INF = -__builtin_inff();

  // -------- stage Q rows m0..m0+15 into LDS (async copy, 16B per issue) ----
  {
    const int tid = threadIdx.x;
    const _Float16* gq = q + (size_t)m0 * DIM;
#pragma unroll
    for (int i = 0; i < 8; ++i) {
      const int c   = tid + i * 256;       // 2048 chunks of 8 f16
      const int row = c >> 7;
      const int col = (c & 127) * 8;
      unsigned lds_addr = (unsigned)(size_t)(void*)&Qs[row * QSTRIDE + col];
      const _Float16* g = gq + (size_t)row * DIM + col;
      asm volatile("global_load_async_to_lds_b128 %0, %1, off"
                   :: "v"(lds_addr), "v"(g) : "memory");
    }
    asm volatile("s_wait_asynccnt 0x0" ::: "memory");
    __syncthreads();
  }
  const _Float16* qrow = Qs + ln * QSTRIDE;   // LDS-resident A rows

  // -------- Phase 1: exact row max / exp-sum over this wave's key tiles ----
  float mrow[8], lrow[8];
#pragma unroll
  for (int r = 0; r < 8; ++r) { mrow[r] = NEG_INF; lrow[r] = 0.f; }

  for (int jt = wave; jt < NTOK / 16; jt += 8) {
    const _Float16* krow = k + (size_t)(jt * 16 + ln) * DIM;
    v8f s = {};
    v16h b = load_b16(krow, 0, half);
#pragma unroll
    for (int kk = 0; kk < DIM; kk += 32) {
      v16h a = load_a16(qrow, kk, half);
      v16h bn = b;
      if (kk + 32 < DIM) bn = load_b16(krow, kk + 32, half);
      s = wmma_f16(a, b, s);
      b = bn;
      __builtin_amdgcn_sched_group_barrier(SG_DSRD,   2, 0);
      __builtin_amdgcn_sched_group_barrier(SG_VMEMRD, 2, 0);
      __builtin_amdgcn_sched_group_barrier(SG_WMMA,   1, 0);
    }
#pragma unroll
    for (int r = 0; r < 8; ++r) {
      float sv = s[r] * INV_SCALE;
      float mx = sv;
      for (int off = 1; off < 16; off <<= 1) mx = fmaxf(mx, __shfl_xor(mx, off, 16));
      float mnew = fmaxf(mrow[r], mx);
      float p = __expf(sv - mnew);
      for (int off = 1; off < 16; off <<= 1) p += __shfl_xor(p, off, 16);
      lrow[r] = lrow[r] * __expf(mrow[r] - mnew) + p;
      mrow[r] = mnew;
    }
  }
  if (ln == 0) {
#pragma unroll
    for (int r = 0; r < 8; ++r) {
      mbuf[wave * 16 + half * 8 + r] = mrow[r];
      lbuf[wave * 16 + half * 8 + r] = lrow[r];
    }
  }
  __syncthreads();

  // combine 8 partial (m,l) per row
  float Mr[8], Li[8];
#pragma unroll
  for (int r = 0; r < 8; ++r) {
    const int row = half * 8 + r;
    float m = NEG_INF;
    for (int w = 0; w < 8; ++w) m = fmaxf(m, mbuf[w * 16 + row]);
    float l = 0.f;
    for (int w = 0; w < 8; ++w) l += lbuf[w * 16 + row] * __expf(mbuf[w * 16 + row] - m);
    Mr[r] = m;
    Li[r] = 1.0f / l;
  }

  // -------- Phase 2: recompute S, write normalized probs to LDS (f16) -----
  for (int jt = wave; jt < NTOK / 16; jt += 8) {
    const int j0 = jt * 16;
    const _Float16* krow = k + (size_t)(j0 + ln) * DIM;
    v8f s = {};
    v16h b = load_b16(krow, 0, half);
#pragma unroll
    for (int kk = 0; kk < DIM; kk += 32) {
      v16h a = load_a16(qrow, kk, half);
      v16h bn = b;
      if (kk + 32 < DIM) bn = load_b16(krow, kk + 32, half);
      s = wmma_f16(a, b, s);
      b = bn;
      __builtin_amdgcn_sched_group_barrier(SG_DSRD,   2, 0);
      __builtin_amdgcn_sched_group_barrier(SG_VMEMRD, 2, 0);
      __builtin_amdgcn_sched_group_barrier(SG_WMMA,   1, 0);
    }
#pragma unroll
    for (int r = 0; r < 8; ++r) {
      float p = __expf(s[r] * INV_SCALE - Mr[r]) * Li[r];
      P[(half * 8 + r) * PSTRIDE + j0 + ln] = (_Float16)p;
    }
  }
  __syncthreads();

  // -------- Phase 3: O = P @ V, wave owns 128 output columns --------------
  const int d0 = wave * 128;
  const _Float16* vrow[8];
#pragma unroll
  for (int t = 0; t < 8; ++t) vrow[t] = vT + (size_t)(d0 + t * 16 + ln) * NTOK;

  v8f acc[8] = {};
  const _Float16* prow = P + ln * PSTRIDE;
#pragma unroll 2
  for (int kk = 0; kk < NTOK; kk += 32) {
    v16h a = load_a16(prow, kk, half);   // LDS
    v16h b[8];
#pragma unroll
    for (int t = 0; t < 8; ++t) b[t] = load_b16(vrow[t], kk, half);   // one clause
#pragma unroll
    for (int t = 0; t < 8; ++t) acc[t] = wmma_f16(a, b[t], acc[t]);
    __builtin_amdgcn_sched_group_barrier(SG_DSRD,    2, 0);
    __builtin_amdgcn_sched_group_barrier(SG_VMEMRD, 16, 0);
    __builtin_amdgcn_sched_group_barrier(SG_WMMA,    8, 0);
  }
#pragma unroll
  for (int t = 0; t < 8; ++t) {
#pragma unroll
    for (int r = 0; r < 8; ++r) {
      o[(size_t)(m0 + half * 8 + r) * DIM + d0 + t * 16 + ln] = (_Float16)acc[t][r];
    }
  }
}

// ---- output projection + bias + residual (fp32 out) ----
__global__ void __launch_bounds__(256)
out_proj(const _Float16* __restrict__ O,    // [NTOK, DIM] f16
         const _Float16* __restrict__ Wo,   // [DIM, DIM] f16
         const float* __restrict__ bo,
         const float* __restrict__ x,
         float* __restrict__ out) {
  const int wave = threadIdx.x >> 5;
  const int lane = threadIdx.x & 31;
  const int half = lane >> 4;
  const int ln   = lane & 15;
  const int gw = blockIdx.x * 8 + wave;
  const int m0 = (gw >> 4) << 4;
  const int d0 = (gw & 15) << 6;

  const _Float16* arow = O + (size_t)(m0 + ln) * DIM;
  const _Float16* wrow[4];
#pragma unroll
  for (int t = 0; t < 4; ++t) wrow[t] = Wo + (size_t)(d0 + t * 16 + ln) * DIM;

  v8f acc[4] = {};
  v16h a = load_a16(arow, 0, half);
  v16h b[4];
#pragma unroll
  for (int t = 0; t < 4; ++t) b[t] = load_b16(wrow[t], 0, half);

#pragma unroll
  for (int kk = 0; kk < DIM; kk += 32) {
    const int kn = kk + 32;
    v16h an = a;
    v16h bn[4] = {b[0], b[1], b[2], b[3]};
    if (kn < DIM) {
      an = load_a16(arow, kn, half);
#pragma unroll
      for (int t = 0; t < 4; ++t) bn[t] = load_b16(wrow[t], kn, half);
    }
#pragma unroll
    for (int t = 0; t < 4; ++t) acc[t] = wmma_f16(a, b[t], acc[t]);
    a = an;
#pragma unroll
    for (int t = 0; t < 4; ++t) b[t] = bn[t];
    __builtin_amdgcn_sched_group_barrier(SG_VMEMRD, 10, 0);
    __builtin_amdgcn_sched_group_barrier(SG_WMMA,    4, 0);
  }

#pragma unroll
  for (int t = 0; t < 4; ++t) {
    const int col = d0 + t * 16 + ln;
    const float bv = bo[col];
#pragma unroll
    for (int r = 0; r < 8; ++r) {
      const int row = m0 + half * 8 + r;
      out[(size_t)row * DIM + col] = acc[t][r] + bv + x[(size_t)row * DIM + col];
    }
  }
}

extern "C" void kernel_launch(void* const* d_in, const int* in_sizes, int n_in,
                              void* d_out, int out_size, void* d_ws, size_t ws_size,
                              hipStream_t stream) {
  const float* x  = (const float*)d_in[0];
  const float* Wq = (const float*)d_in[1];
  const float* bq = (const float*)d_in[2];
  const float* Wk = (const float*)d_in[3];
  const float* bk = (const float*)d_in[4];
  const float* Wv = (const float*)d_in[5];
  const float* bv = (const float*)d_in[6];
  const float* Wo = (const float*)d_in[7];
  const float* bo = (const float*)d_in[8];
  float* out = (float*)d_out;

  // workspace partition (f16), ~88 MB total
  char* ws = (char*)d_ws;
  _Float16* x_h  = (_Float16*)ws; ws += (size_t)NTOK * DIM * 2;
  _Float16* Wq_h = (_Float16*)ws; ws += (size_t)DIM * DIM * 2;
  _Float16* Wk_h = (_Float16*)ws; ws += (size_t)DIM * DIM * 2;
  _Float16* Wv_h = (_Float16*)ws; ws += (size_t)DIM * DIM * 2;
  _Float16* Wo_h = (_Float16*)ws; ws += (size_t)DIM * DIM * 2;
  _Float16* q_h  = (_Float16*)ws; ws += (size_t)NTOK * DIM * 2;
  _Float16* k_h  = (_Float16*)ws; ws += (size_t)NTOK * DIM * 2;
  _Float16* vT_h = (_Float16*)ws; ws += (size_t)NTOK * DIM * 2;
  _Float16* o_h  = (_Float16*)ws; ws += (size_t)NTOK * DIM * 2;

  cvt_f32_f16<<<2048, 256, 0, stream>>>(x,  x_h,  NTOK * DIM);
  cvt_f32_f16<<<1024, 256, 0, stream>>>(Wq, Wq_h, DIM * DIM);
  cvt_f32_f16<<<1024, 256, 0, stream>>>(Wk, Wk_h, DIM * DIM);
  cvt_f32_f16<<<1024, 256, 0, stream>>>(Wv, Wv_h, DIM * DIM);
  cvt_f32_f16<<<1024, 256, 0, stream>>>(Wo, Wo_h, DIM * DIM);

  const dim3 gproj(1024);   // (512 row-tiles * 16 col-groups) / 8 waves
  gemm_proj<<<gproj, 256, 0, stream>>>(x_h, Wq_h, bq, q_h, 0);
  gemm_proj<<<gproj, 256, 0, stream>>>(x_h, Wk_h, bk, k_h, 0);
  gemm_proj<<<gproj, 256, 0, stream>>>(x_h, Wv_h, bv, vT_h, 1);  // -> vT

  const size_t shmem = (size_t)16 * PSTRIDE * 2 +   // P   (~256 KB)
                       (size_t)16 * QSTRIDE * 2 +   // Qs  (~32 KB)
                       2 * 8 * 16 * sizeof(float);  // m/l buffers
  (void)hipFuncSetAttribute((const void*)flash_attn,
                            hipFuncAttributeMaxDynamicSharedMemorySize,
                            (int)shmem);
  flash_attn<<<NTOK / 16, 256, shmem, stream>>>(q_h, k_h, vT_h, o_h);

  out_proj<<<gproj, 256, 0, stream>>>(o_h, Wo_h, bo, x, out);
}
